// BiLSTM_CRF_21277267984674
// MI455X (gfx1250) — compile-verified, matrix operand order checked by reading, not run on hardware
//
#include <hip/hip_runtime.h>

// ---------------------------------------------------------------------------
// BiLSTM-CRF forward loss for MI455X (gfx1250), wave32 + WMMA bf16.
//   V=50000 E=512 H2=256 T=74 B=128 L=256
// GEMM B-operands staged in LDS (64KB / 32KB slabs) to cut L2 re-fetch 8x.
// Load-phase / compute-phase split per K-iter so ds_load clauses pipeline
// ahead of the WMMA chain instead of serializing on s_wait_dscnt 0.
// ---------------------------------------------------------------------------

#define VV 50000
#define EE 512
#define HH 256        // H2
#define G4 1024       // 4*H2
#define TT 74
#define TP 80         // padded tag dim
#define BB 128
#define LL 256
#define MM (LL * BB)  // 32768 rows of the time-batched GEMMs

typedef __attribute__((ext_vector_type(16))) __bf16 v16bf;
typedef __attribute__((ext_vector_type(8)))  float  v8f;

union Frag16 { v16bf v; uint4 u[2]; };

__device__ __forceinline__ v8f zero8() {
  v8f z = {0.f, 0.f, 0.f, 0.f, 0.f, 0.f, 0.f, 0.f};
  return z;
}

// Load one 16x32 bf16 operand fragment (A and B layouts are duals here).
// rowbase must already include row*stride + koff. hi = lane>>4.
// lanes 0-15: K chunks {0..7, 16..23}; lanes 16-31: {8..15, 24..31}.
__device__ __forceinline__ v16bf load_frag(const __bf16* rowbase, int hi) {
  const uint4* p = (const uint4*)(rowbase + 8 * hi);
  Frag16 f;
  f.u[0] = p[0];   // 8 bf16 at koff + 8*hi
  f.u[1] = p[2];   // 8 bf16 at koff + 16 + 8*hi
  return f.v;
}

__device__ __forceinline__ float sigmoidf(float x) {
  return 1.f / (1.f + expf(-x));
}

// ------------------------------- prep kernels ------------------------------

__global__ void k_gather(const int* __restrict__ ids,
                         const float* __restrict__ embed,
                         __bf16* __restrict__ X) {
  size_t gid = (size_t)blockIdx.x * blockDim.x + threadIdx.x;
  if (gid >= (size_t)LL * BB * EE) return;
  int e = (int)(gid & (EE - 1));
  int b = (int)((gid >> 9) & (BB - 1));
  int t = (int)(gid >> 16);
  int id = ids[b * LL + t];
  X[gid] = (__bf16)embed[(size_t)id * EE + e];
}

__global__ void k_f2bf(const float* __restrict__ s, __bf16* __restrict__ d, int n) {
  int i = blockIdx.x * blockDim.x + threadIdx.x;
  if (i < n) d[i] = (__bf16)s[i];
}

__global__ void k_copy_f(const float* __restrict__ s, float* __restrict__ d, int n) {
  int i = blockIdx.x * blockDim.x + threadIdx.x;
  if (i < n) d[i] = s[i];
}

__global__ void k_bias_sum(const float* __restrict__ a, const float* __restrict__ b,
                           float* __restrict__ d, int n) {
  int i = blockIdx.x * blockDim.x + threadIdx.x;
  if (i < n) d[i] = a[i] + b[i];
}

__global__ void k_pad_wout(const float* __restrict__ W, __bf16* __restrict__ D) {
  int gid = blockIdx.x * blockDim.x + threadIdx.x;
  if (gid >= TP * 512) return;
  int r = gid / 512, c = gid % 512;
  D[gid] = (__bf16)((r < TT) ? W[r * 512 + c] : 0.f);
}

__global__ void k_pad_bout(const float* __restrict__ b, float* __restrict__ d) {
  int i = threadIdx.x;
  if (i < TP) d[i] = (i < TT) ? b[i] : 0.f;
}

// --------------------- batched input GEMM: G = X*Wih^T + b ------------------
// X: [MM,512] bf16 row-major; W: [1024,512] bf16 row-major; G: [MM,1024] bf16.
// Block (8 waves) shares one N-group (4 N-tiles = 64 W rows, 64KB) staged in
// LDS once; each wave owns a distinct 16-row M-tile. K loop = 512/32.

__global__ void k_gemm_in(const __bf16* __restrict__ X,
                          const __bf16* __restrict__ Wf, const __bf16* __restrict__ Wb,
                          const float* __restrict__ bf_, const float* __restrict__ bb_,
                          __bf16* __restrict__ Gf, __bf16* __restrict__ Gb) {
  __shared__ __align__(16) __bf16 slab[64 * EE];  // 64KB
  int dir = blockIdx.y;
  const __bf16* W = dir ? Wb : Wf;
  const float* bias = dir ? bb_ : bf_;
  __bf16* G = dir ? Gb : Gf;

  int ng = blockIdx.x & 15;        // N-group (4 tiles: rows ng*64 .. ng*64+63)
  int mgroup = blockIdx.x >> 4;    // 0..255
  int wv = threadIdx.x >> 5;       // 0..7
  int mtile = mgroup * 8 + wv;     // 0..2047
  int lane = threadIdx.x & 31, hi = lane >> 4, lo = lane & 15;

  // stage B slab: 64 rows x 512 bf16, contiguous uint4 copies
  for (int q = threadIdx.x; q < 4096; q += 256) {
    int r = q >> 6, kq = q & 63;
    ((uint4*)slab)[q] =
        *(const uint4*)(W + (size_t)(ng * 64 + r) * EE + kq * 8);
  }
  __syncthreads();

  v8f acc[4];
  for (int j = 0; j < 4; ++j) acc[j] = zero8();

  const __bf16* arow = X + (size_t)(mtile * 16 + lo) * EE;
  for (int kk = 0; kk < 16; ++kk) {
    int koff = kk * 32;
    v16bf a = load_frag(arow + koff, hi);
    // load phase: all 4 B fragments into distinct registers
    Frag16 bf[4];
#pragma unroll
    for (int j = 0; j < 4; ++j) {
      const uint4* p = (const uint4*)(&slab[(j * 16 + lo) * EE + koff + 8 * hi]);
      bf[j].u[0] = p[0];
      bf[j].u[1] = p[2];
    }
    // compute phase: drain with back-to-back WMMAs
#pragma unroll
    for (int j = 0; j < 4; ++j)
      acc[j] = __builtin_amdgcn_wmma_f32_16x16x32_bf16(
          false, a, false, bf[j].v, (short)0, acc[j], false, false);
  }
#pragma unroll
  for (int j = 0; j < 4; ++j) {
    int col = (ng * 4 + j) * 16 + lo;
    float bv = bias[col];
#pragma unroll
    for (int r = 0; r < 8; ++r) {
      int row = mtile * 16 + r + 8 * hi;
      G[(size_t)row * G4 + col] = (__bf16)(acc[j][r] + bv);
    }
  }
}

// --------------------------- fused LSTM step -------------------------------
// g = G[p] + h_prev * Whh^T ; elementwise cell fused into the WMMA epilogue.
// Per direction: block = one hidden tile nh (its 64 Whh rows staged in a
// 32KB LDS slab), 8 waves = 8 batch tiles. Each wave owns the 4 gate
// column-chunks (i,f,g,o) for its 16x16 (b,h) tile so the cell is local.

__global__ void k_step(const __bf16* __restrict__ Gf, const __bf16* __restrict__ Gb,
                       const __bf16* __restrict__ Whhf, const __bf16* __restrict__ Whhb,
                       __bf16* __restrict__ hf, __bf16* __restrict__ hb,
                       float* __restrict__ cf, float* __restrict__ cb, int t) {
  __shared__ __align__(16) __bf16 slabW[64 * HH];  // 32KB
  int dir = blockIdx.z;
  int p = dir ? (LL - 1 - t) : t;
  const __bf16* G = dir ? Gb : Gf;
  const __bf16* Whh = dir ? Whhb : Whhf;
  float* c = dir ? cb : cf;
  const __bf16* hprev = dir ? (hb + (size_t)(p + 1) * BB * HH)
                            : (hf + (size_t)t * BB * HH);
  __bf16* hout = dir ? (hb + (size_t)p * BB * HH)
                     : (hf + (size_t)(t + 1) * BB * HH);

  int nh = blockIdx.x;             // 0..15 hidden tile
  int mtile = threadIdx.x >> 5;    // 0..7 batch tile (one per wave)
  int lane = threadIdx.x & 31, hi = lane >> 4, lo = lane & 15;

  // stage Whh slab: rows (g*16 + c) local <- global row g*256 + nh*16 + c
  for (int q = threadIdx.x; q < 2048; q += 256) {
    int r = q >> 5, kq = q & 31;
    int g = r >> 4, cc = r & 15;
    ((uint4*)slabW)[q] =
        *(const uint4*)(Whh + (size_t)(g * HH + nh * 16 + cc) * HH + kq * 8);
  }
  __syncthreads();

  v8f acc[4];
  for (int g = 0; g < 4; ++g) acc[g] = zero8();

  const __bf16* arow = hprev + (size_t)(mtile * 16 + lo) * HH;
#pragma unroll
  for (int kk = 0; kk < 8; ++kk) {
    int koff = kk * 32;
    v16bf a = load_frag(arow + koff, hi);
    Frag16 bf[4];
#pragma unroll
    for (int g = 0; g < 4; ++g) {
      const uint4* pB = (const uint4*)(&slabW[(g * 16 + lo) * HH + koff + 8 * hi]);
      bf[g].u[0] = pB[0];
      bf[g].u[1] = pB[2];
    }
#pragma unroll
    for (int g = 0; g < 4; ++g)
      acc[g] = __builtin_amdgcn_wmma_f32_16x16x32_bf16(
          false, a, false, bf[g].v, (short)0, acc[g], false, false);
  }
#pragma unroll
  for (int r = 0; r < 8; ++r) {
    int brow = mtile * 16 + r + 8 * hi;
    int hcol = nh * 16 + lo;
    size_t gbase = ((size_t)p * BB + brow) * G4;
    float gi = acc[0][r] + (float)G[gbase + 0 * HH + hcol];
    float gff = acc[1][r] + (float)G[gbase + 1 * HH + hcol];
    float ggg = acc[2][r] + (float)G[gbase + 2 * HH + hcol];
    float go = acc[3][r] + (float)G[gbase + 3 * HH + hcol];
    size_t ci = (size_t)brow * HH + hcol;
    float cn = sigmoidf(gff) * c[ci] + sigmoidf(gi) * tanhf(ggg);
    float hn = sigmoidf(go) * tanhf(cn);
    c[ci] = cn;
    hout[ci] = (__bf16)hn;
  }
}

// ------------------- output projection: em = [hf,hb]*Wout^T + b -------------
// K=512 split: first 256 from hf (slot t+1), last 256 from hb (slot t).
// N padded to 80 (5 tiles); each wave does all 5 N-tiles per A load.

__global__ void k_feats(const __bf16* __restrict__ hf, const __bf16* __restrict__ hb,
                        const __bf16* __restrict__ Wp, const float* __restrict__ bp,
                        float* __restrict__ em) {
  int mtile = blockIdx.x * (blockDim.x >> 5) + (threadIdx.x >> 5);  // 0..2047
  int lane = threadIdx.x & 31, hi = lane >> 4, lo = lane & 15;
  int m = mtile * 16 + lo;
  int t = m >> 7, b = m & (BB - 1);

  v8f acc[5];
  for (int j = 0; j < 5; ++j) acc[j] = zero8();

  for (int kk = 0; kk < 16; ++kk) {
    const __bf16* arow;
    if (kk < 8)
      arow = hf + ((size_t)(t + 1) * BB + b) * HH + kk * 32;
    else
      arow = hb + ((size_t)t * BB + b) * HH + (kk - 8) * 32;
    v16bf a = load_frag(arow, hi);
    int koff = kk * 32;
    Frag16 bf[5];
#pragma unroll
    for (int j = 0; j < 5; ++j) {
      const uint4* p = (const uint4*)(Wp + (size_t)(j * 16 + lo) * 512 + koff + 8 * hi);
      bf[j].u[0] = p[0];
      bf[j].u[1] = p[2];
    }
#pragma unroll
    for (int j = 0; j < 5; ++j)
      acc[j] = __builtin_amdgcn_wmma_f32_16x16x32_bf16(
          false, a, false, bf[j].v, (short)0, acc[j], false, false);
  }
#pragma unroll
  for (int j = 0; j < 5; ++j) {
    int col = j * 16 + lo;
    float bv = bp[col];
#pragma unroll
    for (int r = 0; r < 8; ++r) {
      int row = mtile * 16 + r + 8 * hi;
      em[(size_t)row * TP + col] = acc[j][r] + bv;
    }
  }
}

// ------------------------------- CRF kernel --------------------------------
// One block per batch row. trans in LDS; double-buffered alpha; gold-path
// numerator reduced in parallel over timesteps.

__global__ void k_crf(const float* __restrict__ em, const int* __restrict__ tags,
                      const int* __restrict__ lengths,
                      const float* __restrict__ startt, const float* __restrict__ endt,
                      const float* __restrict__ trans, float* __restrict__ llh) {
  __shared__ float tr[TT * TT];
  __shared__ float a0[TP], a1[TP];
  __shared__ float red[128];
  int b = blockIdx.x, tid = threadIdx.x;

  for (int i = tid; i < TT * TT; i += 128) tr[i] = trans[i];

  int len = lengths[b];
  if (len < 1) len = 1;

  // numerator (gold path score), parallel over timesteps
  float part = 0.f;
  for (int t = 1 + tid; t < len; t += 128) {
    int tp = tags[b * LL + t - 1];
    int tc = tags[b * LL + t];
    part += trans[tp * TT + tc] + em[((size_t)t * BB + b) * TP + tc];
  }
  if (tid == 0) {
    int t0 = tags[b * LL];
    part += startt[t0] + em[(size_t)b * TP + t0];
    part += endt[tags[b * LL + len - 1]];
  }
  red[tid] = part;
  __syncthreads();
  for (int s = 64; s > 0; s >>= 1) {
    if (tid < s) red[tid] += red[tid + s];
    __syncthreads();
  }
  float num = red[0];

  // forward algorithm
  if (tid < TT) a0[tid] = startt[tid] + em[(size_t)b * TP + tid];
  __syncthreads();
  float* cur = a0;
  float* nxt = a1;
  for (int t = 1; t < len; ++t) {
    if (tid < TT) {
      float mx = -3.4e38f;
      for (int i = 0; i < TT; ++i) mx = fmaxf(mx, cur[i] + tr[i * TT + tid]);
      float s = 0.f;
      for (int i = 0; i < TT; ++i) s += expf(cur[i] + tr[i * TT + tid] - mx);
      nxt[tid] = mx + logf(s) + em[((size_t)t * BB + b) * TP + tid];
    }
    __syncthreads();
    float* tmp = cur; cur = nxt; nxt = tmp;
  }

  // den = logsumexp(alpha + end)
  float v = (tid < TT) ? cur[tid] + endt[tid] : -3.4e38f;
  red[tid] = v;
  __syncthreads();
  for (int s = 64; s > 0; s >>= 1) {
    if (tid < s) red[tid] = fmaxf(red[tid], red[tid + s]);
    __syncthreads();
  }
  float mx = red[0];
  __syncthreads();
  red[tid] = (tid < TT) ? expf(v - mx) : 0.f;
  __syncthreads();
  for (int s = 64; s > 0; s >>= 1) {
    if (tid < s) red[tid] += red[tid + s];
    __syncthreads();
  }
  if (tid == 0) llh[b] = num - (mx + logf(red[0]));
}

__global__ void k_reduce(const float* __restrict__ llh, float* __restrict__ out) {
  __shared__ float red[128];
  int tid = threadIdx.x;
  red[tid] = llh[tid];
  __syncthreads();
  for (int s = 64; s > 0; s >>= 1) {
    if (tid < s) red[tid] += red[tid + s];
    __syncthreads();
  }
  if (tid == 0) out[0] = -red[0] / (float)BB;
}

// ------------------------------- launcher ----------------------------------

extern "C" void kernel_launch(void* const* d_in, const int* in_sizes, int n_in,
                              void* d_out, int out_size, void* d_ws, size_t ws_size,
                              hipStream_t stream) {
  (void)in_sizes; (void)n_in; (void)out_size; (void)ws_size;
  const int*   ids    = (const int*)d_in[0];
  const int*   tags   = (const int*)d_in[1];
  const int*   lens   = (const int*)d_in[2];
  const float* embed  = (const float*)d_in[3];
  const float* Wih_f  = (const float*)d_in[4];
  const float* Whh_f  = (const float*)d_in[5];
  const float* bih_f  = (const float*)d_in[6];
  const float* bhh_f  = (const float*)d_in[7];
  const float* Wih_b  = (const float*)d_in[8];
  const float* Whh_b  = (const float*)d_in[9];
  const float* bih_b  = (const float*)d_in[10];
  const float* bhh_b  = (const float*)d_in[11];
  const float* Wout   = (const float*)d_in[12];
  const float* bout   = (const float*)d_in[13];
  const float* startt = (const float*)d_in[14];
  const float* endt   = (const float*)d_in[15];
  const float* trans  = (const float*)d_in[16];
  const float* h0     = (const float*)d_in[17];
  const float* c0     = (const float*)d_in[18];
  float* out = (float*)d_out;

  char* w = (char*)d_ws;
  size_t off = 0;
  auto take = [&](size_t bytes) -> char* {
    char* p = w + off;
    off = (off + bytes + 255) & ~(size_t)255;
    return p;
  };

  __bf16* X      = (__bf16*)take((size_t)MM * EE * 2);        // 33.5 MB
  __bf16* Wihf16 = (__bf16*)take((size_t)G4 * EE * 2);
  __bf16* Wihb16 = (__bf16*)take((size_t)G4 * EE * 2);
  __bf16* Gf     = (__bf16*)take((size_t)MM * G4 * 2);        // 67 MB
  __bf16* Gb     = (__bf16*)take((size_t)MM * G4 * 2);        // 67 MB
  __bf16* Whhf16 = (__bf16*)take((size_t)G4 * HH * 2);
  __bf16* Whhb16 = (__bf16*)take((size_t)G4 * HH * 2);
  __bf16* Woutp  = (__bf16*)take((size_t)TP * 512 * 2);
  float*  bsumf  = (float*)take((size_t)G4 * 4);
  float*  bsumb  = (float*)take((size_t)G4 * 4);
  float*  boutp  = (float*)take((size_t)TP * 4);
  __bf16* hf     = (__bf16*)take((size_t)(LL + 1) * BB * HH * 2);  // 16.8 MB
  __bf16* hb     = (__bf16*)take((size_t)(LL + 1) * BB * HH * 2);  // 16.8 MB
  float*  cf     = (float*)take((size_t)BB * HH * 4);
  float*  cb     = (float*)take((size_t)BB * HH * 4);
  float*  em     = (float*)take((size_t)MM * TP * 4);         // 10.5 MB
  float*  llh    = (float*)take((size_t)BB * 4);

  // ---- prep: gather + precision conversion + initial states ----
  k_gather<<<(MM * EE + 255) / 256, 256, 0, stream>>>(ids, embed, X);
  k_f2bf<<<(G4 * EE + 255) / 256, 256, 0, stream>>>(Wih_f, Wihf16, G4 * EE);
  k_f2bf<<<(G4 * EE + 255) / 256, 256, 0, stream>>>(Wih_b, Wihb16, G4 * EE);
  k_f2bf<<<(G4 * HH + 255) / 256, 256, 0, stream>>>(Whh_f, Whhf16, G4 * HH);
  k_f2bf<<<(G4 * HH + 255) / 256, 256, 0, stream>>>(Whh_b, Whhb16, G4 * HH);
  k_bias_sum<<<(G4 + 255) / 256, 256, 0, stream>>>(bih_f, bhh_f, bsumf, G4);
  k_bias_sum<<<(G4 + 255) / 256, 256, 0, stream>>>(bih_b, bhh_b, bsumb, G4);
  k_pad_wout<<<(TP * 512 + 255) / 256, 256, 0, stream>>>(Wout, Woutp);
  k_pad_bout<<<1, 128, 0, stream>>>(bout, boutp);
  // h0[0] -> forward slot 0, h0[1] -> backward slot L
  k_f2bf<<<(BB * HH + 255) / 256, 256, 0, stream>>>(h0, hf, BB * HH);
  k_f2bf<<<(BB * HH + 255) / 256, 256, 0, stream>>>(h0 + BB * HH,
                                                    hb + (size_t)LL * BB * HH, BB * HH);
  k_copy_f<<<(BB * HH + 255) / 256, 256, 0, stream>>>(c0, cf, BB * HH);
  k_copy_f<<<(BB * HH + 255) / 256, 256, 0, stream>>>(c0 + BB * HH, cb, BB * HH);

  // ---- hoisted time-batched input GEMM (both directions) ----
  // blockIdx.x = {mgroup(0..255), ng(0..15)}; 8 waves/block share one LDS
  // B-slab and cover 8 M-tiles.
  k_gemm_in<<<dim3(4096, 2), 256, 0, stream>>>(X, Wihf16, Wihb16, bsumf, bsumb, Gf, Gb);

  // ---- sequential recurrence: 256 fused steps, both directions per launch ----
  // block = one hidden tile (LDS Whh slab), 8 waves = batch tiles.
  for (int t = 0; t < LL; ++t)
    k_step<<<dim3(16, 1, 2), 256, 0, stream>>>(Gf, Gb, Whhf16, Whhb16, hf, hb, cf, cb, t);

  // ---- output projection ----
  k_feats<<<256, 256, 0, stream>>>(hf, hb, Woutp, boutp, em);

  // ---- CRF + reduction ----
  k_crf<<<BB, 128, 0, stream>>>(em, tags, lens, startt, endt, trans, llh);
  k_reduce<<<1, 128, 0, stream>>>(llh, out);
}
